// GAT_70334384439325
// MI455X (gfx1250) — matrix operand order repaired and checked by reading, not compiled, not run on hardware
//
#include <hip/hip_runtime.h>

typedef __attribute__((ext_vector_type(16))) _Float16 v16h;
typedef __attribute__((ext_vector_type(8)))  _Float16 v8h;
typedef __attribute__((ext_vector_type(8)))  float    v8f;

#define NNODES 10000
#define NEDGES 80000
#define NHEADS 4
#define DIN    128
#define DD1    512
#define DD2    32
#define DOUTF  64
#define NEG_SLOPE 0.2f

// ---------------------------------------------------------------- utilities

__global__ void fill_kernel(float* __restrict__ p, float v, long n) {
    long i = (long)blockIdx.x * blockDim.x + threadIdx.x;
    if (i < n) p[i] = v;
}

// f32 -> f16 elementwise
__global__ void convert_f16_kernel(const float* __restrict__ a,
                                   _Float16* __restrict__ o, long n) {
    long i = (long)blockIdx.x * blockDim.x + threadIdx.x;
    if (i < n) o[i] = (_Float16)a[i];
}

// W[K,N] (row-major) -> Wt[N,K] f16 (transposed + converted)
__global__ void convert_transpose_kernel(const float* __restrict__ W,
                                         _Float16* __restrict__ Wt, int K, int N) {
    long i = (long)blockIdx.x * blockDim.x + threadIdx.x;
    if (i >= (long)N * K) return;
    int n = (int)(i / K), k = (int)(i % K);
    Wt[i] = (_Float16)W[(long)k * N + n];
}

__device__ inline void atomicMaxFloat(float* addr, float v) {
    // sign-split: int-max for non-negative, uint-min for negative; init = -inf.
    if (v >= 0.0f) atomicMax((int*)addr, __float_as_int(v));
    else           atomicMin((unsigned int*)addr, __float_as_uint(v));
}

__device__ inline v16h combine16(v8h lo, v8h hi) {
    v16h r;
    #pragma unroll
    for (int j = 0; j < 8; ++j) { r[j] = lo[j]; r[8 + j] = hi[j]; }
    return r;
}

// ------------------------------------------------------------- WMMA GEMM
// C[M,N] = Ah[M,K](f16) @ Wt[N,K](f16, pre-transposed) + bias[N], f32 out.
// Requires M%16==0, N%64==0, K%32==0. One wave32 per 16x64 output tile
// (4 accumulators share one A fragment); 8 waves per block. The k-loop is
// software-pipelined: next chunk's 10 b128 loads are in flight while the
// current chunk's 4 WMMAs execute.
__global__ __launch_bounds__(256)
void gemm_wmma_kernel(const _Float16* __restrict__ Ah, const _Float16* __restrict__ Wt,
                      const float* __restrict__ bias, float* __restrict__ C,
                      int M, int K, int N) {
    const int tilesN = N >> 6;
    const int nTiles = (M >> 4) * tilesN;
    const int tile   = blockIdx.x * 8 + (threadIdx.x >> 5);  // wave-uniform
    if (tile >= nTiles) return;                               // whole-wave exit

    const int tm   = (tile / tilesN) << 4;
    const int tn   = (tile % tilesN) << 6;
    const int lane = threadIdx.x & 31;
    const int sel  = lane >> 4;   // half-wave selector
    const int rc   = lane & 15;   // row (A) / col (B,C)

    // A: lane holds row tm+rc; halves 0-7 -> K=k0+8*sel+j, 8-15 -> +16
    const _Float16* pa = Ah + (long)(tm + rc) * K + 8 * sel;
    // B: lane holds col; halves 0-7 -> K=k0+8*sel+j, 8-15 -> +16
    const _Float16* pb = Wt + (long)(tn + rc) * K + 8 * sel;
    const long bs = (long)16 * K;   // stride to next 16-column group

    auto ldA = [&](int k0) -> v16h {
        return combine16(*(const v8h*)(pa + k0), *(const v8h*)(pa + k0 + 16));
    };
    auto ldB = [&](int k0, int t) -> v16h {
        const _Float16* p = pb + (long)t * bs + k0;
        return combine16(*(const v8h*)p, *(const v8h*)(p + 16));
    };

    v8f acc0 = {}, acc1 = {}, acc2 = {}, acc3 = {};

    // prologue: fragments for chunk 0
    v16h a  = ldA(0);
    v16h b0 = ldB(0, 0), b1 = ldB(0, 1), b2 = ldB(0, 2), b3 = ldB(0, 3);

    for (int k0 = 32; k0 < K; k0 += 32) {
        // issue next chunk's loads before consuming current fragments
        v16h an  = ldA(k0);
        v16h b0n = ldB(k0, 0), b1n = ldB(k0, 1), b2n = ldB(k0, 2), b3n = ldB(k0, 3);
        acc0 = __builtin_amdgcn_wmma_f32_16x16x32_f16(false, a, false, b0, (short)0, acc0, false, false);
        acc1 = __builtin_amdgcn_wmma_f32_16x16x32_f16(false, a, false, b1, (short)0, acc1, false, false);
        acc2 = __builtin_amdgcn_wmma_f32_16x16x32_f16(false, a, false, b2, (short)0, acc2, false, false);
        acc3 = __builtin_amdgcn_wmma_f32_16x16x32_f16(false, a, false, b3, (short)0, acc3, false, false);
        a = an; b0 = b0n; b1 = b1n; b2 = b2n; b3 = b3n;
    }
    // epilogue: last chunk
    acc0 = __builtin_amdgcn_wmma_f32_16x16x32_f16(false, a, false, b0, (short)0, acc0, false, false);
    acc1 = __builtin_amdgcn_wmma_f32_16x16x32_f16(false, a, false, b1, (short)0, acc1, false, false);
    acc2 = __builtin_amdgcn_wmma_f32_16x16x32_f16(false, a, false, b2, (short)0, acc2, false, false);
    acc3 = __builtin_amdgcn_wmma_f32_16x16x32_f16(false, a, false, b3, (short)0, acc3, false, false);

    const float* pbias = bias + tn + rc;
    const float bn0 = pbias[0], bn1 = pbias[16], bn2 = pbias[32], bn3 = pbias[48];
    float* pc = C + (long)(tm + 8 * sel) * N + tn + rc;   // row = tm+8*sel+r, col = tn+16t+rc
    #pragma unroll
    for (int r = 0; r < 8; ++r) {
        pc[(long)r * N +  0] = acc0[r] + bn0;
        pc[(long)r * N + 16] = acc1[r] + bn1;
        pc[(long)r * N + 32] = acc2[r] + bn2;
        pc[(long)r * N + 48] = acc3[r] + bn3;
    }
}

// ------------------------------------------------------------- edge kernels

__global__ __launch_bounds__(256)
void edge_score_kernel(const float* __restrict__ fs, const float* __restrict__ fd,
                       const float* __restrict__ attn, const int* __restrict__ src,
                       const int* __restrict__ dst, float* __restrict__ eout,
                       float* __restrict__ mout, int D) {
    const int gw = (blockIdx.x * 256 + (int)threadIdx.x) >> 5;  // wave-uniform
    if (gw >= NEDGES * NHEADS) return;
    const int lane = threadIdx.x & 31;
    const int e = gw >> 2;        // NHEADS == 4
    const int h = gw & 3;
    const int s = src[e], d = dst[e];
    const float* pfs = fs + ((long)s * NHEADS + h) * D;
    const float* pfd = fd + ((long)d * NHEADS + h) * D;
    const float* pat = attn + h * D;
    float acc = 0.0f;
    for (int i = lane; i < D; i += 32) {
        float v = pfs[i] + pfd[i];
        v = v > 0.0f ? v : NEG_SLOPE * v;
        acc += pat[i] * v;
    }
    #pragma unroll
    for (int off = 16; off; off >>= 1) acc += __shfl_xor(acc, off, 32);
    if (lane == 0) {
        eout[gw] = acc;
        atomicMaxFloat(mout + d * NHEADS + h, acc);
    }
}

__global__ void edge_exp_kernel(float* __restrict__ ebuf, const float* __restrict__ m,
                                const int* __restrict__ dst, float* __restrict__ ssum) {
    const int g = blockIdx.x * 256 + threadIdx.x;
    if (g >= NEDGES * NHEADS) return;
    const int e = g >> 2, h = g & 3;
    const int d = dst[e];
    const float ex = expf(ebuf[g] - m[d * NHEADS + h]);
    ebuf[g] = ex;
    atomicAdd(ssum + d * NHEADS + h, ex);
}

__global__ void edge_aggregate_kernel(const float* __restrict__ fs, const float* __restrict__ ex,
                                      const float* __restrict__ ssum, const int* __restrict__ src,
                                      const int* __restrict__ dst, float* __restrict__ rst, int D) {
    const long idx = (long)blockIdx.x * 256 + threadIdx.x;
    const long total = (long)NEDGES * NHEADS * D;
    if (idx >= total) return;
    const int HD = NHEADS * D;
    const int e = (int)(idx / HD);
    const int r = (int)(idx % HD);
    const int h = r / D;
    const int s = src[e], d = dst[e];
    const float alpha = ex[e * NHEADS + h] / ssum[d * NHEADS + h];
    atomicAdd(rst + (long)d * HD + r, fs[(long)s * HD + r] * alpha);
}

// o(f16) = relu(a + b); output feeds the next GEMM so convert inline
__global__ void relu_add_f16_kernel(const float* __restrict__ a, const float* __restrict__ b,
                                    _Float16* __restrict__ o, long n) {
    const long i = (long)blockIdx.x * 256 + threadIdx.x;
    if (i < n) {
        const float v = a[i] + b[i];
        o[i] = (_Float16)(v > 0.0f ? v : 0.0f);
    }
}

// ---------------------------------------------------------------- launcher

extern "C" void kernel_launch(void* const* d_in, const int* in_sizes, int n_in,
                              void* d_out, int out_size, void* d_ws, size_t ws_size,
                              hipStream_t stream) {
    const float* features = (const float*)d_in[0];
    const int*   src      = (const int*)  d_in[1];
    const int*   dst      = (const int*)  d_in[2];
    const float* W1s=(const float*)d_in[3],  *b1s=(const float*)d_in[4];
    const float* W1d=(const float*)d_in[5],  *b1d=(const float*)d_in[6];
    const float* attn1=(const float*)d_in[7];
    const float* W1r=(const float*)d_in[8],  *b1r=(const float*)d_in[9];
    const float* W2s=(const float*)d_in[10], *b2s=(const float*)d_in[11];
    const float* W2d=(const float*)d_in[12], *b2d=(const float*)d_in[13];
    const float* attn2=(const float*)d_in[14];
    const float* W2r=(const float*)d_in[15], *b2r=(const float*)d_in[16];
    const float* Wout=(const float*)d_in[17],*bout=(const float*)d_in[18];

    const long NF1 = (long)NNODES * NHEADS * DD1;   // 20,480,000
    const long NF2 = (long)NNODES * NHEADS * DD2;   //  1,280,000
    const long EH  = (long)NEDGES * NHEADS;         //    320,000
    const long NHn = (long)NNODES * NHEADS;         //     40,000
    const long WSZ = (long)DIN * NHEADS * DD1;      //    262,144 (all big weights)

    float* wsf  = (float*)d_ws;
    float* bufA = wsf;              // fs1 (f32), later h1 (f16)
    float* bufB = bufA + NF1;       // fd1, later rst1, later layer-2 arena
    float* bufC = bufB + NF1;       // res1, later h2 (f16)
    float* eb1  = bufC + NF1;       // edge scores / exp
    float* m1   = eb1 + EH;
    float* s1   = m1 + NHn;
    // half-precision arena (weights + converted features)
    _Float16* featH = (_Float16*)(s1 + NHn);        // NNODES*DIN
    _Float16* Wt1s  = featH + (long)NNODES * DIN;
    _Float16* Wt1d  = Wt1s + WSZ;
    _Float16* Wt1r  = Wt1d + WSZ;
    _Float16* Wt2s  = Wt1r + WSZ;
    _Float16* Wt2d  = Wt2s + WSZ;
    _Float16* Wt2r  = Wt2d + WSZ;
    _Float16* WtO   = Wt2r + WSZ;                   // 64*128
    // layer-2 arena carved from bufB (dead after layer-1 relu)
    float* fs2  = bufB;
    float* fd2  = fs2 + NF2;
    float* res2 = fd2 + NF2;
    float* rst2 = res2 + NF2;
    float* eb2  = rst2 + NF2;
    float* m2   = eb2 + EH;
    float* s2   = m2 + NHn;
    _Float16* h1h = (_Float16*)bufA;                // layer-1 output, f16
    _Float16* h2h = (_Float16*)bufC;                // layer-2 output, f16

    const float NEG_INF = -__builtin_inff();

    auto gemm = [&](const _Float16* A, const _Float16* Wt, const float* b, float* C,
                    int M, int K, int N) {
        const int nTiles = (M / 16) * (N / 64);
        const int blocks = (nTiles + 7) / 8;
        gemm_wmma_kernel<<<blocks, 256, 0, stream>>>(A, Wt, b, C, M, K, N);
    };
    auto fill = [&](float* p, float v, long n) {
        fill_kernel<<<(unsigned)((n + 255) / 256), 256, 0, stream>>>(p, v, n);
    };
    auto xpose = [&](const float* W, _Float16* Wt, int K, int N) {
        long n = (long)K * N;
        convert_transpose_kernel<<<(unsigned)((n + 255) / 256), 256, 0, stream>>>(W, Wt, K, N);
    };

    // ---------------- precision prep (once per launch) ----------------
    convert_f16_kernel<<<(unsigned)(((long)NNODES * DIN + 255) / 256), 256, 0, stream>>>(
        features, featH, (long)NNODES * DIN);
    xpose(W1s, Wt1s, DIN, NHEADS * DD1);
    xpose(W1d, Wt1d, DIN, NHEADS * DD1);
    xpose(W1r, Wt1r, DIN, NHEADS * DD1);
    xpose(W2s, Wt2s, NHEADS * DD1, NHEADS * DD2);
    xpose(W2d, Wt2d, NHEADS * DD1, NHEADS * DD2);
    xpose(W2r, Wt2r, NHEADS * DD1, NHEADS * DD2);
    xpose(Wout, WtO, NHEADS * DD2, DOUTF);

    // ---------------- layer 1 (D = 512) ----------------
    gemm(featH, Wt1s, b1s, bufA, NNODES, DIN, NHEADS * DD1);   // fs1
    gemm(featH, Wt1d, b1d, bufB, NNODES, DIN, NHEADS * DD1);   // fd1
    gemm(featH, Wt1r, b1r, bufC, NNODES, DIN, NHEADS * DD1);   // res1
    fill(m1, NEG_INF, NHn);
    edge_score_kernel<<<(unsigned)(EH / 8), 256, 0, stream>>>(
        bufA, bufB, attn1, src, dst, eb1, m1, DD1);
    fill(s1, 0.0f, NHn);
    edge_exp_kernel<<<(unsigned)((EH + 255) / 256), 256, 0, stream>>>(eb1, m1, dst, s1);
    fill(bufB, 0.0f, NF1);                                     // rst1 (fd1 dead)
    edge_aggregate_kernel<<<(unsigned)((EH * DD1 + 255) / 256), 256, 0, stream>>>(
        bufA, eb1, s1, src, dst, bufB, DD1);
    relu_add_f16_kernel<<<(unsigned)((NF1 + 255) / 256), 256, 0, stream>>>(
        bufB, bufC, h1h, NF1);                                 // h1 = relu(rst1+res1), f16

    // ---------------- layer 2 (D = 32) ----------------
    gemm(h1h, Wt2s, b2s, fs2, NNODES, NHEADS * DD1, NHEADS * DD2);
    gemm(h1h, Wt2d, b2d, fd2, NNODES, NHEADS * DD1, NHEADS * DD2);
    gemm(h1h, Wt2r, b2r, res2, NNODES, NHEADS * DD1, NHEADS * DD2);
    fill(m2, NEG_INF, NHn);
    edge_score_kernel<<<(unsigned)(EH / 8), 256, 0, stream>>>(
        fs2, fd2, attn2, src, dst, eb2, m2, DD2);
    fill(s2, 0.0f, NHn);
    edge_exp_kernel<<<(unsigned)((EH + 255) / 256), 256, 0, stream>>>(eb2, m2, dst, s2);
    fill(rst2, 0.0f, NF2);
    edge_aggregate_kernel<<<(unsigned)((EH * DD2 + 255) / 256), 256, 0, stream>>>(
        fs2, eb2, s2, src, dst, rst2, DD2);
    relu_add_f16_kernel<<<(unsigned)((NF2 + 255) / 256), 256, 0, stream>>>(
        rst2, res2, h2h, NF2);

    // ---------------- output projection ----------------
    gemm(h2h, WtO, bout, (float*)d_out, NNODES, NHEADS * DD2, DOUTF);
}
